// GPT2Attention_73246372266220
// MI455X (gfx1250) — matrix operand order repaired and checked by reading, not compiled, not run on hardware
//
#include <hip/hip_runtime.h>
#include <hip/hip_bf16.h>

#define SEQ    2048
#define PREV   2048
#define HIDDEN 768
#define NHEAD  12
#define HDIM   64
#define TOTKV  (PREV + SEQ)   // 4096
#define N3H    (3 * HIDDEN)   // 2304

typedef __bf16 bf16;
typedef bf16  v16bf __attribute__((ext_vector_type(16)));
typedef bf16  v8bf  __attribute__((ext_vector_type(8)));
typedef float v8f   __attribute__((ext_vector_type(8)));
typedef unsigned int v4u __attribute__((ext_vector_type(4)));
typedef int   v8i   __attribute__((ext_vector_type(8)));
typedef int   v4i   __attribute__((ext_vector_type(4)));

// ---- TDM availability (device pass only; host pass parses fallback) -------
#if defined(__gfx1250__) && __has_builtin(__builtin_amdgcn_tensor_load_to_lds)
#define HAVE_TDM 1
#if defined(__has_include)
#if __has_include(<hip/amd_detail/amd_gfx1250_TDM.h>)
#define TDM_SIX_ARGS 1
#endif
#endif
#else
#define HAVE_TDM 0
#endif

union AB16 { v16bf v; struct { v8bf lo, hi; } p; };

static __device__ __forceinline__ v8f wmma_bf16(v16bf a, v16bf b, v8f c) {
  // D = A(16x32 bf16) * B(32x16 bf16) + C(16x16 f32)
  return __builtin_amdgcn_wmma_f32_16x16x32_bf16(false, a, false, b, (short)0, c,
                                                 false, false);
}

// A-fragment (16x32, M x K), row-major source with leading dim `ld`.
static __device__ __forceinline__ v16bf load_a_frag(const bf16* __restrict__ base,
                                                    int ld, int m, int k0, int lane) {
  const int kb = k0 + ((lane >> 4) << 4);
  const bf16* p = base + (size_t)(m + (lane & 15)) * ld + kb;
  AB16 u;
  u.p.lo = *(const v8bf*)(p);
  u.p.hi = *(const v8bf*)(p + 8);
  return u.v;
}

static __device__ __forceinline__ void wait_tensorcnt0() {
#if defined(__gfx1250__)
#if __has_builtin(__builtin_amdgcn_s_wait_tensorcnt)
  __builtin_amdgcn_s_wait_tensorcnt(0);
#else
  asm volatile("s_wait_tensorcnt 0x0" ::: "memory");
#endif
#endif
}

#if HAVE_TDM
// Issue a TDM 2-D tile load (bf16 elements) into LDS.
// tensor: rows of tdim0 elements, tdim1 rows, row stride stride0 (elements).
// tile:   tiled0 x tiled1 (elements x rows), packed contiguously into LDS.
static __device__ __forceinline__ void tdm_load_2d(unsigned int lds_addr,
                                                   const bf16* gaddr,
                                                   unsigned int tdim0, unsigned int tdim1,
                                                   unsigned int tiled0, unsigned int tiled1,
                                                   unsigned long long stride0) {
  unsigned long long ga = (unsigned long long)(uintptr_t)gaddr;
  v4u g0;
  g0[0] = 1u;                                   // count=1, user descriptor
  g0[1] = lds_addr;                             // LDS byte address
  g0[2] = (unsigned int)ga;                     // global_addr[31:0]
  g0[3] = (unsigned int)((ga >> 32) & 0x01FFFFFFull) | 0x80000000u; // addr[56:32], type=2
  v8i g1;
  g1[0] = (int)(1u << 16);                      // data_size=1 (2 bytes)
  g1[1] = (int)((tdim0 & 0xFFFFu) << 16);       // tensor_dim0 lo16 (bits 63:48)
  g1[2] = (int)((tdim0 >> 16) | ((tdim1 & 0xFFFFu) << 16)); // dim0 hi16 | dim1 lo16
  g1[3] = (int)((tdim1 >> 16) | (tiled0 << 16));            // dim1 hi16 | tile_dim0
  g1[4] = (int)tiled1;                          // tile_dim1 (tile_dim2 = 0)
  g1[5] = (int)(unsigned int)(stride0 & 0xFFFFFFFFull);     // dim0_stride[31:0]
  g1[6] = (int)(unsigned int)((stride0 >> 32) & 0xFFFFull); // dim0_stride[47:32]
  g1[7] = 0;                                    // tensor_dim1_stride (unused, 2D)
  v4i z4 = {0, 0, 0, 0};
#if defined(TDM_SIX_ARGS)
  v8i z8 = {0, 0, 0, 0, 0, 0, 0, 0};
  __builtin_amdgcn_tensor_load_to_lds(g0, g1, z4, z4, z8, 0);
#else
  __builtin_amdgcn_tensor_load_to_lds(g0, g1, z4, z4, 0);
#endif
}
#endif

// ---------------------------------------------------------------------------
// Prep kernels
// ---------------------------------------------------------------------------
__global__ void k_cvt_bf16(const float* __restrict__ src, bf16* __restrict__ dst, int n) {
  int i = blockIdx.x * 256 + threadIdx.x;
  if (i < n) dst[i] = (bf16)src[i];
}

// Pack W[K][N] (row major fp32) into B-fragment layout: pack[(k/16)*N + n][16] = W[k][n]
__global__ void k_pack_w(const float* __restrict__ w, bf16* __restrict__ pack, int K, int N) {
  int i = blockIdx.x * 256 + threadIdx.x;
  if (i >= K * N) return;
  int k = i / N, n = i - k * N;
  pack[((((size_t)(k >> 4)) * N + n) << 4) + (k & 15)] = (bf16)w[i];
}

// last_key [h][PREV][d] -> kall [h][TOTKV][d] (first PREV rows)
__global__ void k_cvt_lastkey(const float* __restrict__ lk, bf16* __restrict__ kall) {
  int i = blockIdx.x * 256 + threadIdx.x;
  if (i >= NHEAD * PREV * HDIM) return;
  int d = i & (HDIM - 1);
  int t = i >> 6;
  int j = t & (PREV - 1);
  int h = t >> 11;
  kall[((size_t)h * TOTKV + j) * HDIM + d] = (bf16)lk[i];
}

// last_value [h][PREV][d] -> vallT [h][d][TOTKV] (first PREV cols), transposed
__global__ void k_cvt_lastval(const float* __restrict__ lv, bf16* __restrict__ vallT) {
  int i = blockIdx.x * 256 + threadIdx.x;
  if (i >= NHEAD * PREV * HDIM) return;
  int d = i & (HDIM - 1);
  int t = i >> 6;
  int j = t & (PREV - 1);
  int h = t >> 11;
  vallT[((size_t)h * HDIM + d) * TOTKV + j] = (bf16)lv[i];
}

// ---------------------------------------------------------------------------
// QKV GEMM: [2048,768] x [768,2304] + bias, scatter epilogue
// ---------------------------------------------------------------------------
__global__ __launch_bounds__(256) void k_qkv(
    const bf16* __restrict__ Xb, const bf16* __restrict__ Wp,
    const float* __restrict__ bias,
    bf16* __restrict__ q_ws, bf16* __restrict__ kall, bf16* __restrict__ vallT,
    float* __restrict__ outK, float* __restrict__ outV) {
  const int lane  = threadIdx.x & 31;
  const int wave  = threadIdx.x >> 5;
  const int m0    = blockIdx.x * 128 + (wave >> 1) * 32;
  const int n0    = blockIdx.y * 128 + (wave & 1) * 64;
  const int nlane = lane & 15;
  const int half  = lane >> 4;

  const v8f vzero = {0.f, 0.f, 0.f, 0.f, 0.f, 0.f, 0.f, 0.f};
  v8f acc[2][4];
#pragma unroll
  for (int mi = 0; mi < 2; ++mi)
#pragma unroll
    for (int ni = 0; ni < 4; ++ni) acc[mi][ni] = vzero;

  for (int k0 = 0; k0 < HIDDEN; k0 += 32) {
    v16bf a0 = load_a_frag(Xb, HIDDEN, m0, k0, lane);
    v16bf a1 = load_a_frag(Xb, HIDDEN, m0 + 16, k0, lane);
    const int kt = (k0 >> 4) + half;
#pragma unroll
    for (int ni = 0; ni < 4; ++ni) {
      const int n = n0 + ni * 16 + nlane;
      v16bf b = *(const v16bf*)(Wp + (((size_t)kt * N3H + n) << 4));
      acc[0][ni] = wmma_bf16(a0, b, acc[0][ni]);
      acc[1][ni] = wmma_bf16(a1, b, acc[1][ni]);
    }
  }

#pragma unroll
  for (int mi = 0; mi < 2; ++mi) {
#pragma unroll
    for (int ni = 0; ni < 4; ++ni) {
      const int c = n0 + ni * 16 + nlane;
      const float bc = bias[c];
#pragma unroll
      for (int r = 0; r < 8; ++r) {
        const int s = m0 + mi * 16 + half * 8 + r;
        const float v = acc[mi][ni][r] + bc;
        if (c < HIDDEN) {                      // Q (pre-scaled by 1/sqrt(dh))
          const int h = c >> 6, d = c & 63;
          q_ws[((size_t)h * SEQ + s) * HDIM + d] = (bf16)(v * 0.125f);
        } else if (c < 2 * HIDDEN) {           // K
          const int cc = c - HIDDEN, h = cc >> 6, d = cc & 63;
          outK[((size_t)h * SEQ + s) * HDIM + d] = v;
          kall[((size_t)h * TOTKV + PREV + s) * HDIM + d] = (bf16)v;
        } else {                               // V
          const int cc = c - 2 * HIDDEN, h = cc >> 6, d = cc & 63;
          outV[((size_t)h * SEQ + s) * HDIM + d] = v;
          vallT[((size_t)h * HDIM + d) * TOTKV + PREV + s] = (bf16)v;
        }
      }
    }
  }
}

// ---------------------------------------------------------------------------
// Flash attention: block = (head, 128 queries), 8 waves x 16 rows each.
// K/V tiles staged in LDS via TDM (double-buffered); online softmax;
// P transposed C->A layout via LDS.
// ---------------------------------------------------------------------------
__global__ __launch_bounds__(256) void k_attn(
    const bf16* __restrict__ q_ws, const bf16* __restrict__ kall,
    const bf16* __restrict__ vallT, bf16* __restrict__ ctx_b) {
  __shared__ bf16 ldsP[8][16][32];
#if HAVE_TDM
  __shared__ bf16 kbuf[2][32 * HDIM];   // [key][d] per tile
  __shared__ bf16 vbuf[2][HDIM * 32];   // [d][key] per tile
#endif

  const int lane  = threadIdx.x & 31;
  const int wave  = threadIdx.x >> 5;
  const int nlane = lane & 15;
  const int half  = lane >> 4;
  const int h     = blockIdx.x;
  const int qs0   = blockIdx.y * 128;
  const int qbase = qs0 + wave * 16;

  const bf16* qh = q_ws  + (size_t)h * SEQ   * HDIM;
  const bf16* kh = kall  + (size_t)h * TOTKV * HDIM;
  const bf16* vh = vallT + (size_t)h * HDIM  * TOTKV;

  const v16bf qa0 = load_a_frag(qh, HDIM, qbase, 0,  lane);
  const v16bf qa1 = load_a_frag(qh, HDIM, qbase, 32, lane);

  const v8f vzero = {0.f, 0.f, 0.f, 0.f, 0.f, 0.f, 0.f, 0.f};
  v8f acc[4];
#pragma unroll
  for (int f = 0; f < 4; ++f) acc[f] = vzero;
  float mrun[8], lrun[8];
#pragma unroll
  for (int r = 0; r < 8; ++r) { mrun[r] = -1e30f; lrun[r] = 0.f; }

  const int jmax   = PREV + qs0 + 128;   // exclusive key bound for this block
  const int ntiles = jmax >> 5;

#if HAVE_TDM
  // Prefetch tile 0 (wave 0 drives the TDM; TENSORcnt is per-wave).
  if (wave == 0) {
    tdm_load_2d((unsigned int)(uintptr_t)&kbuf[0][0], kh, HDIM, TOTKV, HDIM, 32, HDIM);
    tdm_load_2d((unsigned int)(uintptr_t)&vbuf[0][0], vh, TOTKV, HDIM, 32, HDIM, TOTKV);
    wait_tensorcnt0();
  }
  __syncthreads();
#endif

  for (int t = 0; t < ntiles; ++t) {
    const int jt  = t << 5;
#if HAVE_TDM
    const int buf = t & 1;
    if (wave == 0 && t + 1 < ntiles) {      // async prefetch of next tile
      const int jn = jt + 32;
      tdm_load_2d((unsigned int)(uintptr_t)&kbuf[buf ^ 1][0], kh + (size_t)jn * HDIM,
                  HDIM, TOTKV, HDIM, 32, HDIM);
      tdm_load_2d((unsigned int)(uintptr_t)&vbuf[buf ^ 1][0], vh + jn,
                  TOTKV, HDIM, 32, HDIM, TOTKV);
    }
    const bf16* kt0 = &kbuf[buf][0];
    v16bf b0lo = *(const v16bf*)(kt0 + nlane * HDIM + half * 16);
    v16bf b0hi = *(const v16bf*)(kt0 + nlane * HDIM + 32 + half * 16);
    v16bf b1lo = *(const v16bf*)(kt0 + (nlane + 16) * HDIM + half * 16);
    v16bf b1hi = *(const v16bf*)(kt0 + (nlane + 16) * HDIM + 32 + half * 16);
#else
    const bf16* kr0 = kh + (size_t)(jt + nlane) * HDIM;
    const bf16* kr1 = kh + (size_t)(jt + 16 + nlane) * HDIM;
    v16bf b0lo = *(const v16bf*)(kr0 + half * 16);
    v16bf b0hi = *(const v16bf*)(kr0 + 32 + half * 16);
    v16bf b1lo = *(const v16bf*)(kr1 + half * 16);
    v16bf b1hi = *(const v16bf*)(kr1 + 32 + half * 16);
#endif
    v8f s0 = wmma_bf16(qa0, b0lo, wmma_bf16(qa1, b0hi, vzero));
    v8f s1 = wmma_bf16(qa0, b1lo, wmma_bf16(qa1, b1hi, vzero));

    const bool domask = (jt + 31 > PREV + qbase);

    // ---- online softmax (rows m = half*8 + r, replicated over 16 lanes) ----
#pragma unroll
    for (int r = 0; r < 8; ++r) {
      const int m = half * 8 + r;
      float v0 = s0[r], v1 = s1[r];
      if (domask) {
        const int lim = PREV + qbase + m;
        if (jt + nlane > lim)       v0 = -1e30f;
        if (jt + 16 + nlane > lim)  v1 = -1e30f;
      }
      float tmax = fmaxf(v0, v1);
      tmax = fmaxf(tmax, __shfl_xor(tmax, 1, 32));
      tmax = fmaxf(tmax, __shfl_xor(tmax, 2, 32));
      tmax = fmaxf(tmax, __shfl_xor(tmax, 4, 32));
      tmax = fmaxf(tmax, __shfl_xor(tmax, 8, 32));
      const float mnew = fmaxf(mrun[r], tmax);
      const float sc   = __expf(mrun[r] - mnew);
      mrun[r] = mnew;
      const float p0 = __expf(v0 - mnew);
      const float p1 = __expf(v1 - mnew);
      float rs = p0 + p1;
      rs += __shfl_xor(rs, 1, 32);
      rs += __shfl_xor(rs, 2, 32);
      rs += __shfl_xor(rs, 4, 32);
      rs += __shfl_xor(rs, 8, 32);
      lrun[r] = lrun[r] * sc + rs;
#pragma unroll
      for (int f = 0; f < 4; ++f) acc[f][r] *= sc;
      ldsP[wave][m][nlane]      = (bf16)p0;
      ldsP[wave][m][nlane + 16] = (bf16)p1;
    }
    __syncthreads();

    // ---- ctx += P (16x32) x V (32x64) ----
    {
      AB16 u;
      u.p.lo = *(const v8bf*)&ldsP[wave][nlane][half * 16];
      u.p.hi = *(const v8bf*)&ldsP[wave][nlane][half * 16 + 8];
      const v16bf pa = u.v;
#pragma unroll
      for (int f = 0; f < 4; ++f) {
#if HAVE_TDM
        v16bf vb = *(const v16bf*)(&vbuf[buf][0] + (f * 16 + nlane) * 32 + half * 16);
#else
        v16bf vb = *(const v16bf*)(vh + (size_t)(f * 16 + nlane) * TOTKV + jt + half * 16);
#endif
        acc[f] = wmma_bf16(pa, vb, acc[f]);
      }
    }
#if HAVE_TDM
    if (wave == 0) wait_tensorcnt0();   // next tile resident before anyone reads it
#endif
    __syncthreads();
  }

  // ---- normalize and write ctx (bf16, seq-major [S, HIDDEN]) ----
#pragma unroll
  for (int f = 0; f < 4; ++f) {
#pragma unroll
    for (int r = 0; r < 8; ++r) {
      const int m = half * 8 + r;
      const int s = qbase + m;
      const int col = h * HDIM + f * 16 + nlane;
      ctx_b[(size_t)s * HIDDEN + col] = (bf16)(acc[f][r] / lrun[r]);
    }
  }
}

// ---------------------------------------------------------------------------
// Output projection: ctx [2048,768] x Wproj [768,768] + bias -> d_out
// ---------------------------------------------------------------------------
__global__ __launch_bounds__(256) void k_proj(
    const bf16* __restrict__ Cb, const bf16* __restrict__ Wp,
    const float* __restrict__ bias, float* __restrict__ out) {
  const int lane  = threadIdx.x & 31;
  const int wave  = threadIdx.x >> 5;
  const int m0    = blockIdx.x * 128 + (wave >> 1) * 32;
  const int n0    = blockIdx.y * 128 + (wave & 1) * 64;
  const int nlane = lane & 15;
  const int half  = lane >> 4;

  const v8f vzero = {0.f, 0.f, 0.f, 0.f, 0.f, 0.f, 0.f, 0.f};
  v8f acc[2][4];
#pragma unroll
  for (int mi = 0; mi < 2; ++mi)
#pragma unroll
    for (int ni = 0; ni < 4; ++ni) acc[mi][ni] = vzero;

  for (int k0 = 0; k0 < HIDDEN; k0 += 32) {
    v16bf a0 = load_a_frag(Cb, HIDDEN, m0, k0, lane);
    v16bf a1 = load_a_frag(Cb, HIDDEN, m0 + 16, k0, lane);
    const int kt = (k0 >> 4) + half;
#pragma unroll
    for (int ni = 0; ni < 4; ++ni) {
      const int n = n0 + ni * 16 + nlane;
      v16bf b = *(const v16bf*)(Wp + (((size_t)kt * HIDDEN + n) << 4));
      acc[0][ni] = wmma_bf16(a0, b, acc[0][ni]);
      acc[1][ni] = wmma_bf16(a1, b, acc[1][ni]);
    }
  }

#pragma unroll
  for (int mi = 0; mi < 2; ++mi)
#pragma unroll
    for (int ni = 0; ni < 4; ++ni) {
      const int c = n0 + ni * 16 + nlane;
      const float bc = bias[c];
#pragma unroll
      for (int r = 0; r < 8; ++r) {
        const int s = m0 + mi * 16 + half * 8 + r;
        out[(size_t)s * HIDDEN + c] = acc[mi][ni][r] + bc;
      }
    }
}

// ---------------------------------------------------------------------------
// Launcher
// ---------------------------------------------------------------------------
extern "C" void kernel_launch(void* const* d_in, const int* in_sizes, int n_in,
                              void* d_out, int out_size, void* d_ws, size_t ws_size,
                              hipStream_t stream) {
  const float* hs  = (const float*)d_in[0];   // [2048, 768]
  const float* lk  = (const float*)d_in[1];   // [12, 2048, 64]
  const float* lv  = (const float*)d_in[2];   // [12, 2048, 64]
  const float* wa  = (const float*)d_in[3];   // [768, 2304]
  const float* ba  = (const float*)d_in[4];   // [2304]
  const float* wp  = (const float*)d_in[5];   // [768, 768]
  const float* bp  = (const float*)d_in[6];   // [768]

  float* outO = (float*)d_out;                         // [2048, 768]
  float* outK = outO + (size_t)SEQ * HIDDEN;           // [12, 2048, 64]
  float* outV = outK + (size_t)NHEAD * SEQ * HDIM;     // [12, 2048, 64]

  size_t off = 0;
  auto wsAlloc = [&](size_t bytes) -> void* {
    void* p = (char*)d_ws + off;
    off += (bytes + 255) & ~(size_t)255;
    return p;
  };
  bf16* Xb    = (bf16*)wsAlloc((size_t)SEQ * HIDDEN * 2);
  bf16* Wap   = (bf16*)wsAlloc((size_t)HIDDEN * N3H * 2);
  bf16* Wpp   = (bf16*)wsAlloc((size_t)HIDDEN * HIDDEN * 2);
  bf16* q_ws  = (bf16*)wsAlloc((size_t)NHEAD * SEQ * HDIM * 2);
  bf16* kall  = (bf16*)wsAlloc((size_t)NHEAD * TOTKV * HDIM * 2);
  bf16* vallT = (bf16*)wsAlloc((size_t)NHEAD * HDIM * TOTKV * 2);
  bf16* ctxb  = (bf16*)wsAlloc((size_t)SEQ * HIDDEN * 2);

  const int nX  = SEQ * HIDDEN;
  const int nKV = NHEAD * PREV * HDIM;
  k_cvt_bf16   <<<(nX + 255) / 256, 256, 0, stream>>>(hs, Xb, nX);
  k_pack_w     <<<(HIDDEN * N3H + 255) / 256, 256, 0, stream>>>(wa, Wap, HIDDEN, N3H);
  k_pack_w     <<<(HIDDEN * HIDDEN + 255) / 256, 256, 0, stream>>>(wp, Wpp, HIDDEN, HIDDEN);
  k_cvt_lastkey<<<(nKV + 255) / 256, 256, 0, stream>>>(lk, kall);
  k_cvt_lastval<<<(nKV + 255) / 256, 256, 0, stream>>>(lv, vallT);

  k_qkv <<<dim3(SEQ / 128, N3H / 128), 256, 0, stream>>>(Xb, Wap, ba, q_ws, kall,
                                                         vallT, outK, outV);
  k_attn<<<dim3(NHEAD, SEQ / 128), 256, 0, stream>>>(q_ws, kall, vallT, ctxb);
  k_proj<<<dim3(SEQ / 128, HIDDEN / 128), 256, 0, stream>>>(ctxb, Wpp, bp, outO);
}
